// DDPMVAEQueryEncoder_16879221473941
// MI455X (gfx1250) — matrix operand order repaired and checked by reading, not compiled, not run on hardware
//
#include <hip/hip_runtime.h>
#include <hip/hip_bf16.h>
#include <math.h>

#define TSTEPS 50
#define D      64
#define H4     256
#define BATCH  4096
#define SEQL   200
#define WV     4          // waves per sampler block; 4*16 = 64 rows/block

typedef __attribute__((ext_vector_type(16))) __bf16 v16bf;
typedef __attribute__((ext_vector_type(8)))  float  v8f;

// Workspace layout (32-bit units):
//   [0      .. 8191 ]  W1 B-fragments (bf16-pair packed u32)
//   [8192   .. 16383]  W2 B-fragments
//   [16384  .. 16575]  schedule: Ac[50] +0, Bc[50] +64, sigma[50] +128
//   [16704  .. 19903]  temb[50][64] f32
//   [19968  .. 282111] cond c[4096][64] f32
#define WS_SCHED 16384
#define WS_TEMB  16704
#define WS_COND  19968

__device__ __forceinline__ unsigned short f2bf(float f) {
  unsigned u = __float_as_uint(f);
  unsigned r = (u + 0x7FFFu + ((u >> 16) & 1u)) >> 16;   // RNE
  return (unsigned short)r;
}

// ---------------------------------------------------------------------------
// Kernel 0: diffusion schedule (double precision, matches numpy) + temb table.
// The x-update is folded:  x' = Ac*x - Bc*eps + sg*noise
//   Ac = coef1*sqrt_recip + coef2,  Bc = coef1*sqrt_recipm1
// ---------------------------------------------------------------------------
__global__ void sched_temb_kernel(const float* __restrict__ Wt,
                                  const float* __restrict__ bt,
                                  float* __restrict__ ws) {
  __shared__ double beta_s[TSTEPS], acp_s[TSTEPS];
  int tid = threadIdx.x;
  if (tid == 0) {
    double scale = 1000.0 / (double)TSTEPS;
    double b0 = scale * 1e-4, bN = scale * 2e-2;
    double prod = 1.0;
    for (int i = 0; i < TSTEPS; ++i) {
      double b = b0 + (bN - b0) * (double)i / (double)(TSTEPS - 1);
      beta_s[i] = b;
      prod *= (1.0 - b);
      acp_s[i] = prod;
    }
  }
  __syncthreads();
  float* Ac = ws + WS_SCHED;
  float* Bc = Ac + 64;
  float* sg = Ac + 128;
  if (tid < TSTEPS) {
    int t = tid;
    double a  = acp_s[t];
    double ap = (t == 0) ? 1.0 : acp_s[t - 1];
    double b  = beta_s[t];
    double sqrec   = sqrt(1.0 / a);
    double sqrecm1 = sqrt(1.0 / a - 1.0);
    double co1 = b * sqrt(ap) / (1.0 - a);
    double co2 = (1.0 - ap) * sqrt(1.0 - b) / (1.0 - a);
    Ac[t] = (float)(co1 * sqrec + co2);
    Bc[t] = (float)(co1 * sqrecm1);
    double pv = b * (1.0 - ap) / (1.0 - a);
    sg[t] = (t == 0) ? 0.0f : (float)sqrt(pv);
  }
  __syncthreads();
  // temb[t][n] = [cos(t*f), sin(t*f)] @ Wt + bt   (t is batch-uniform!)
  float* temb = ws + WS_TEMB;
  for (int i = tid; i < TSTEPS * D; i += blockDim.x) {
    int t = i / D, n = i % D;
    float acc = bt[n];
    for (int h = 0; h < 32; ++h) {
      float fr  = __expf(-logf(10000.0f) * (float)h / 32.0f);
      float arg = (float)t * fr;
      acc += __cosf(arg) * Wt[h * D + n] + __sinf(arg) * Wt[(32 + h) * D + n];
    }
    temb[i] = acc;
  }
}

// ---------------------------------------------------------------------------
// Kernel 1: pre-swizzle W1/W2 into WMMA B-fragment layout (bf16 pairs in u32)
// B 32x16 bf16 layout: lane group g, VGPR v holds K = (v>>2)*16 + g*8 + 2*(v&3) (+0,+1)
// ---------------------------------------------------------------------------
__global__ void pack_weights_kernel(const float* __restrict__ W1,
                                    const float* __restrict__ W2,
                                    unsigned* __restrict__ ws) {
  int i = blockIdx.x * blockDim.x + threadIdx.x;   // 0..16383
  int j    = i & 8191;
  int fr   = j >> 8;
  int lane = (j >> 3) & 31;
  int v    = j & 7;
  int g    = lane >> 4;
  if (i < 8192) {                                  // W1 [64 x 256] K-major
    int nt = fr >> 1, ks = fr & 1;
    int n = nt * 16 + (lane & 15);
    int k = ks * 32 + (v >> 2) * 16 + g * 8 + 2 * (v & 3);
    unsigned lo = f2bf(W1[k * H4 + n]);
    unsigned hi = f2bf(W1[(k + 1) * H4 + n]);
    ws[i] = lo | (hi << 16);
  } else {                                         // W2 [256 x 64]
    int nt = fr >> 3, ks = fr & 7;
    int n = nt * 16 + (lane & 15);
    int k = ks * 32 + (v >> 2) * 16 + g * 8 + 2 * (v & 3);
    unsigned lo = f2bf(W2[k * D + n]);
    unsigned hi = f2bf(W2[(k + 1) * D + n]);
    ws[i] = lo | (hi << 16);
  }
}

// ---------------------------------------------------------------------------
// Kernel 2: gather-pool + encoder + cond projection.  One wave per batch row.
// cond c = (pooled @ W_enc[:, :64] + b_enc[:64]) @ Wc + bc   (loop-invariant)
// ---------------------------------------------------------------------------
__global__ void encoder_kernel(const int* __restrict__ seq,
                               const float* __restrict__ item_emb,
                               const float* __restrict__ W_enc,
                               const float* __restrict__ b_enc,
                               const float* __restrict__ Wc,
                               const float* __restrict__ bc,
                               float* __restrict__ ws) {
  __shared__ float pooled_s[8][D];
  __shared__ float mu_s[8][D];
  int lane = threadIdx.x & 31;
  int wv   = threadIdx.x >> 5;
  int b    = blockIdx.x * 8 + wv;
  int n0   = 2 * lane;

  float a0 = 0.f, a1 = 0.f;
  int nnz = 0;
  const int* srow = seq + (size_t)b * SEQL;
  for (int l = 0; l < SEQL; ++l) {
    int idx = srow[l];                 // lane-uniform -> broadcast load
    nnz += (idx != 0);
    const float* e = item_emb + (size_t)idx * D + n0;
    a0 += e[0];
    a1 += e[1];
  }
  float inv = rsqrtf((float)nnz);
  pooled_s[wv][n0]     = a0 * inv;
  pooled_s[wv][n0 + 1] = a1 * inv;
  __syncthreads();

  float m0 = b_enc[n0], m1 = b_enc[n0 + 1];
  #pragma unroll 4
  for (int k = 0; k < D; ++k) {
    float pv = pooled_s[wv][k];
    m0 += pv * W_enc[k * (2 * D) + n0];
    m1 += pv * W_enc[k * (2 * D) + n0 + 1];
  }
  mu_s[wv][n0]     = m0;
  mu_s[wv][n0 + 1] = m1;
  __syncthreads();

  float c0 = bc[n0], c1 = bc[n0 + 1];
  #pragma unroll 4
  for (int k = 0; k < D; ++k) {
    float mv = mu_s[wv][k];
    c0 += mv * Wc[k * D + n0];
    c1 += mv * Wc[k * D + n0 + 1];
  }
  float* cond = ws + WS_COND;
  cond[(size_t)b * D + n0]     = c0;
  cond[(size_t)b * D + n0 + 1] = c1;
}

// ---------------------------------------------------------------------------
// Kernel 3: persistent 50-step DDPM sampler. One wave = 16 rows; x lives in
// registers in the WMMA C-layout (M=(lane/16)*8+r, N=lane%16) for all steps.
// Noise is cacheline-prefetched (global_prefetch_b8) at step start and loaded
// only at the update, so no VGPRs are pinned across the GEMMs.
// ---------------------------------------------------------------------------
__global__ __launch_bounds__(WV * 32, 1)
void sampler_kernel(const float* __restrict__ init_noise,
                    const float* __restrict__ step_noise,
                    const float* __restrict__ b1,
                    const float* __restrict__ b2,
                    const unsigned* __restrict__ wfrag,
                    const float* __restrict__ wsf,
                    float* __restrict__ out) {
  __shared__ __align__(32) unsigned xaf[WV][2][32][8];   // Xa A-fragments (bf16)
  __shared__ __align__(32) unsigned hf[WV][8][32][8];    // H  A-fragments (bf16)

  int lane  = threadIdx.x & 31;
  int wv    = threadIdx.x >> 5;
  int cn    = lane & 15;           // C-layout column within tile
  int gsel  = cn >> 3;             // A-layout dest lane-group for this column
  int vvp   = (cn & 7) >> 1;       // A-layout dest vgpr (within half)
  int p     = cn & 1;              // A-layout 16-bit half
  int mrow0 = (lane >> 4) * 8;     // C-layout row base: m = mrow0 + r
  int rowbase = blockIdx.x * (WV * 16) + wv * 16;

  const float* Ac_a = wsf + WS_SCHED;
  const float* Bc_a = Ac_a + 64;
  const float* sg_a = Ac_a + 128;
  const float* temb = wsf + WS_TEMB;
  const float* cond = wsf + WS_COND;

  float x[4][8], c[4][8];
  #pragma unroll
  for (int nt = 0; nt < 4; ++nt)
    #pragma unroll
    for (int r = 0; r < 8; ++r) {
      size_t off = (size_t)(rowbase + mrow0 + r) * D + nt * 16 + cn;
      x[nt][r] = init_noise[off];
      c[nt][r] = cond[off];
    }
  float b1r[16], b2r[4];
  #pragma unroll
  for (int n1 = 0; n1 < 16; ++n1) b1r[n1] = b1[n1 * 16 + cn];
  #pragma unroll
  for (int nt = 0; nt < 4; ++nt) b2r[nt] = b2[nt * 16 + cn];

  const v16bf* w1f = (const v16bf*)wfrag;           // frag stride = 32 lanes
  const v16bf* w2f = (const v16bf*)(wfrag + 8192);
  const v8f vzero = {0.f, 0.f, 0.f, 0.f, 0.f, 0.f, 0.f, 0.f};

  for (int t = TSTEPS - 1; t >= 0; --t) {
    float Acs = Ac_a[t], Bcs = Bc_a[t], sgs = sg_a[t];
    float tv[4];
    #pragma unroll
    for (int nt = 0; nt < 4; ++nt) tv[nt] = temb[t * D + nt * 16 + cn];

    // Prefetch this step's gaussian noise rows into cache (no VGPR cost).
    // Each 256B row = two 128B lines: prefetch at +0B (covers nt 0,1) and
    // +128B (covers nt 2,3); lanes 0-15 / 16-31 cover rows m and m+8.
    const float* nzbase = step_noise + (size_t)t * (BATCH * D);
    #pragma unroll
    for (int r = 0; r < 8; ++r) {
      const float* pr = nzbase + (size_t)(rowbase + mrow0 + r) * D + cn;
      __builtin_prefetch(pr, 0, 3);
      __builtin_prefetch(pr + 32, 0, 3);
    }

    // Stage Xa = x + temb + c into A-fragment layout (transpose via LDS).
    #pragma unroll
    for (int nt = 0; nt < 4; ++nt) {
      int f = nt >> 1, v = (nt & 1) * 4 + vvp;
      #pragma unroll
      for (int r = 0; r < 8; ++r) {
        int dlane = gsel * 16 + mrow0 + r;
        unsigned short* dst = (unsigned short*)&xaf[wv][f][dlane][v];
        dst[p] = f2bf(x[nt][r] + tv[nt] + c[nt][r]);
      }
    }
    v16bf A0 = *(const v16bf*)&xaf[wv][0][lane][0];
    v16bf A1 = *(const v16bf*)&xaf[wv][1][lane][0];

    // GEMM1: H = silu(Xa @ W1 + b1), written to LDS in A-fragment layout.
    #pragma unroll
    for (int nt1 = 0; nt1 < 16; ++nt1) {
      v8f acc = vzero;
      v16bf Bm0 = w1f[(nt1 * 2 + 0) * 32 + lane];
      v16bf Bm1 = w1f[(nt1 * 2 + 1) * 32 + lane];
      acc = __builtin_amdgcn_wmma_f32_16x16x32_bf16(false, A0, false, Bm0,
                                                    (short)0, acc, false, false);
      acc = __builtin_amdgcn_wmma_f32_16x16x32_bf16(false, A1, false, Bm1,
                                                    (short)0, acc, false, false);
      int f = nt1 >> 1, v = (nt1 & 1) * 4 + vvp;
      #pragma unroll
      for (int r = 0; r < 8; ++r) {
        float h = acc[r] + b1r[nt1];
        // silu via fast reciprocal (v_rcp_f32), not IEEE division
        float s = h * __builtin_amdgcn_rcpf(1.0f + __expf(-h));
        int dlane = gsel * 16 + mrow0 + r;
        unsigned short* dst = (unsigned short*)&hf[wv][f][dlane][v];
        dst[p] = f2bf(s);
      }
    }

    // GEMM2: eps = H @ W2 + b2  (K=256 -> 8 WMMA k-steps, 4 N-tiles).
    v8f acc2[4];
    #pragma unroll
    for (int nt = 0; nt < 4; ++nt) acc2[nt] = vzero;
    #pragma unroll
    for (int ks = 0; ks < 8; ++ks) {
      v16bf HA = *(const v16bf*)&hf[wv][ks][lane][0];
      #pragma unroll
      for (int nt = 0; nt < 4; ++nt) {
        v16bf Bm = w2f[(nt * 8 + ks) * 32 + lane];
        acc2[nt] = __builtin_amdgcn_wmma_f32_16x16x32_bf16(false, HA, false, Bm,
                                                (short)0, acc2[nt], false, false);
      }
    }

    // Folded ancestral DDPM update: x' = Ac*x - Bc*eps + sg*noise.
    // Noise loads hit prefetched lines (WGP$/L2).
    #pragma unroll
    for (int nt = 0; nt < 4; ++nt)
      #pragma unroll
      for (int r = 0; r < 8; ++r) {
        float nzv = nzbase[(size_t)(rowbase + mrow0 + r) * D + nt * 16 + cn];
        float eps = acc2[nt][r] + b2r[nt];
        x[nt][r] = Acs * x[nt][r] - Bcs * eps + sgs * nzv;
      }
  }

  #pragma unroll
  for (int nt = 0; nt < 4; ++nt)
    #pragma unroll
    for (int r = 0; r < 8; ++r)
      out[(size_t)(rowbase + mrow0 + r) * D + nt * 16 + cn] = x[nt][r];
}

// ---------------------------------------------------------------------------
extern "C" void kernel_launch(void* const* d_in, const int* in_sizes, int n_in,
                              void* d_out, int out_size, void* d_ws, size_t ws_size,
                              hipStream_t stream) {
  const int*   seq        = (const int*)  d_in[0];
  const float* item_emb   = (const float*)d_in[1];
  const float* W_enc      = (const float*)d_in[2];
  const float* b_enc      = (const float*)d_in[3];
  const float* Wt         = (const float*)d_in[4];
  const float* bt         = (const float*)d_in[5];
  const float* Wc         = (const float*)d_in[6];
  const float* bc         = (const float*)d_in[7];
  const float* W1         = (const float*)d_in[8];
  const float* b1         = (const float*)d_in[9];
  const float* W2         = (const float*)d_in[10];
  const float* b2         = (const float*)d_in[11];
  const float* init_noise = (const float*)d_in[12];
  const float* step_noise = (const float*)d_in[13];
  float*    out = (float*)   d_out;
  float*    wsf = (float*)   d_ws;
  unsigned* wsu = (unsigned*)d_ws;

  sched_temb_kernel<<<1, 256, 0, stream>>>(Wt, bt, wsf);
  pack_weights_kernel<<<64, 256, 0, stream>>>(W1, W2, wsu);
  encoder_kernel<<<BATCH / 8, 256, 0, stream>>>(seq, item_emb, W_enc, b_enc,
                                                Wc, bc, wsf);
  sampler_kernel<<<BATCH / (WV * 16), WV * 32, 0, stream>>>(
      init_noise, step_noise, b1, b2, wsu, wsf, out);
}
